// BiEncoder_64639257805370
// MI455X (gfx1250) — matrix operand order repaired and checked by reading, not compiled, not run on hardware
//
#include <hip/hip_runtime.h>
#include <stdint.h>
#include <stddef.h>

// Problem constants (from reference): V,E,H,L,B
#define V_ 32000
#define E_ 512
#define H_ 1024
#define L_ 512
#define B_ 64

typedef __bf16 bf16_t;
typedef __bf16 v16bf __attribute__((ext_vector_type(16)));
typedef __bf16 v8bf  __attribute__((ext_vector_type(8)));
typedef float  v8f   __attribute__((ext_vector_type(8)));
typedef unsigned int v4u __attribute__((ext_vector_type(4)));
typedef int          v4i __attribute__((ext_vector_type(4)));
typedef int          v8i __attribute__((ext_vector_type(8)));

// LDS slab geometry: TDM padding inserts 4 dwords (8 bf16) after every
// 256 dwords (512 bf16) -> effective chunk stride 520 elements.
#define LDS_CHUNK 512
#define LDS_CPAD  520
#define WSLAB 8320    /* 16 rows x 512 cols ->  16 chunks * 520 */
#define USLAB 16640   /* 16 rows x 1024 cols -> 32 chunks * 520 */

// Build the 16-element bf16 operand from two 16-byte (8 x bf16) chunks.
// Uses contiguous k order (not the ISA table order; A and B agree):
//  A 16x32:  a[i] = A[m][kb + (lane/16)*16 + i]  (contiguous in k)
//  B 32x16:  b[i] = B[kb + (lane/16)*16 + i][n]  (contiguous in k)
static __device__ __forceinline__ v16bf ld2x8(const bf16_t* p0, const bf16_t* p1) {
  v8bf lo = *(const v8bf*)p0;
  v8bf hi = *(const v8bf*)p1;
  return __builtin_shufflevector(lo, hi, 0,1,2,3,4,5,6,7,8,9,10,11,12,13,14,15);
}

#define WMMA_BF16(ACC, A, Bv)                                                  \
  (ACC) = __builtin_amdgcn_wmma_f32_16x16x32_bf16(false, (A), false, (Bv),     \
                                                  (short)0, (ACC), false, false)

// ---------------------------------------------------------------------------
// Tensor Data Mover: DMA a 2D bf16 tile (rows x cols, row stride in elements)
// from global memory into LDS at byte offset lds_off, with per-512-element
// padding of 16 bytes (bank-conflict avoidance for the WMMA B-operand reads).
// D# bitfields per CDNA5 ISA.
// ---------------------------------------------------------------------------
static __device__ __forceinline__ void tdm_load_2d(const bf16_t* gptr,
                                                   unsigned lds_off,
                                                   int rows, int cols,
                                                   int row_stride) {
  unsigned long long ga = (unsigned long long)(uintptr_t)gptr;
  v4u g0;
  g0.x = 1u;                                              // count=1 (valid user D#)
  g0.y = lds_off;                                         // lds_addr (bytes)
  g0.z = (unsigned)(ga & 0xffffffffu);                    // global_addr[31:0]
  g0.w = (unsigned)((ga >> 32) & 0x1ffffffu) | (2u << 30);// addr[56:32] | type=2
  v8i g1;
  // data_size=1 (2B) | pad_enable | pad_interval=7 (256 dw) | pad_amount=3 (4 dw)
  g1[0] = (1 << 16) | (1 << 20) | (7 << 22) | (3 << 25);
  g1[1] = (cols & 0xffff) << 16;                          // tensor_dim0[15:0]
  g1[2] = ((cols >> 16) & 0xffff) | ((rows & 0xffff) << 16); // dim0 hi | dim1 lo
  g1[3] = ((rows >> 16) & 0xffff) | ((cols & 0xffff) << 16); // dim1 hi | tile_dim0
  g1[4] = rows & 0xffff;                                  // tile_dim1 (tile_dim2=0)
  g1[5] = row_stride;                                     // tensor_dim0_stride[31:0]
  g1[6] = 0;
  g1[7] = 0;
  v4i z4 = {0, 0, 0, 0};
#if __clang_major__ >= 23
  v8i z8 = {0, 0, 0, 0, 0, 0, 0, 0};
  __builtin_amdgcn_tensor_load_to_lds(g0, g1, z4, z4, z8, 0);
#else
  __builtin_amdgcn_tensor_load_to_lds(g0, g1, z4, z4, 0);
#endif
}

// Map a linear (row*cols + k) element index into the padded LDS slab layout.
static __device__ __forceinline__ int lds_idx(int linear) {
  return (linear >> 9) * LDS_CPAD + (linear & 511);
}

// ---------------------------------------------------------------------------
// Small utility kernels
// ---------------------------------------------------------------------------
__global__ void cast_f32_bf16(const float* __restrict__ src,
                              bf16_t* __restrict__ dst, int n) {
  int i = blockIdx.x * blockDim.x + threadIdx.x;
  if (i < n) dst[i] = (bf16_t)src[i];
}

__global__ void zero_f32(float* __restrict__ p, size_t n) {
  size_t i = (size_t)blockIdx.x * blockDim.x + threadIdx.x;
  if (i < n) p[i] = 0.0f;
}

__global__ void zero_bf16(bf16_t* __restrict__ p, size_t n) {
  size_t i = (size_t)blockIdx.x * blockDim.x + threadIdx.x;
  if (i < n) p[i] = (bf16_t)0.0f;
}

// Embedding gather + f32->bf16 cast for both directions.
__global__ void gather_embed(const float* __restrict__ emb,
                             const int* __restrict__ tokens,
                             const int* __restrict__ lengths,
                             bf16_t* __restrict__ x_ltr,
                             bf16_t* __restrict__ x_rtl) {
  size_t i = (size_t)blockIdx.x * blockDim.x + threadIdx.x;
  const size_t total = (size_t)L_ * B_ * E_;
  if (i >= total) return;
  const int e = (int)(i % E_);
  const int b = (int)((i / E_) % B_);
  const int t = (int)(i / ((size_t)E_ * B_));
  const int tok_l = tokens[t * B_ + b];
  x_ltr[i] = (bf16_t)emb[(size_t)tok_l * E_ + e];
  const int ridx  = lengths[b] - 1 - t;
  const int tok_r = (ridx >= 0) ? tokens[ridx * B_ + b] : 0;
  x_rtl[i] = (bf16_t)emb[(size_t)tok_r * E_ + e];
}

// ---------------------------------------------------------------------------
// One GRU timestep, both directions. grid = (64, 2), block = 128 (4 waves).
// A block owns one 16-wide N-tile; its 4 waves own the 4 M(batch)-tiles and
// SHARE the weight B-tiles, which are staged into LDS by the TDM.
// ---------------------------------------------------------------------------
struct StepArgs {
  const bf16_t* x[2];
  const bf16_t* Wr[2]; const bf16_t* Wz[2]; const bf16_t* Wh[2];
  const bf16_t* Ur[2]; const bf16_t* Uz[2]; const bf16_t* Uh[2];
  const float*  br[2]; const float*  bz[2]; const float*  bh[2];
  const float*  hin_f[2];  const bf16_t* hin_b[2];
  float*        hout_f[2]; bf16_t*       hout_b[2];
  const int*    len;
  float*        out;   // hiddens [L, B, 2H]
  int           t;
};

__global__ __launch_bounds__(128) void bigru_step_wmma(StepArgs p) {
  __shared__ __align__(16) bf16_t lds[3 * WSLAB + 3 * USLAB];  // ~146 KB

  const int dir   = blockIdx.y;
  const int wave  = threadIdx.x >> 5;      // M-tile (batch rows wave*16..+15)
  const int lane  = threadIdx.x & 31;
  const int nbase = blockIdx.x * 16;       // N-tile (64 blocks cover H)
  const int mbase = wave * 16;
  const int hsel  = lane >> 4;             // which K-half this lane holds
  const int l16   = lane & 15;
  const int n     = nbase + l16;

  // --- Stage the block's weight slabs into LDS with the Tensor Data Mover ---
  if (threadIdx.x < 32) {                  // one TDM op per descriptor, wave 0
    const unsigned lb = (unsigned)(uintptr_t)&lds[0];
    tdm_load_2d(p.Wr[dir] + (size_t)nbase * E_, lb + 0u * WSLAB * 2u, 16, E_, E_);
    tdm_load_2d(p.Wz[dir] + (size_t)nbase * E_, lb + 1u * WSLAB * 2u, 16, E_, E_);
    tdm_load_2d(p.Wh[dir] + (size_t)nbase * E_, lb + 2u * WSLAB * 2u, 16, E_, E_);
    tdm_load_2d(p.Ur[dir] + (size_t)nbase * H_, lb + (3u * WSLAB + 0u * USLAB) * 2u, 16, H_, H_);
    tdm_load_2d(p.Uz[dir] + (size_t)nbase * H_, lb + (3u * WSLAB + 1u * USLAB) * 2u, 16, H_, H_);
    tdm_load_2d(p.Uh[dir] + (size_t)nbase * H_, lb + (3u * WSLAB + 2u * USLAB) * 2u, 16, H_, H_);
    __builtin_amdgcn_s_wait_tensorcnt(0);
  }
  __syncthreads();

  const bf16_t* wrS = &lds[0 * WSLAB];
  const bf16_t* wzS = &lds[1 * WSLAB];
  const bf16_t* whS = &lds[2 * WSLAB];
  const bf16_t* urS = &lds[3 * WSLAB + 0 * USLAB];
  const bf16_t* uzS = &lds[3 * WSLAB + 1 * USLAB];
  const bf16_t* uhS = &lds[3 * WSLAB + 2 * USLAB];

  v8f acc_r = {}, acc_z = {}, acc_xh = {}, acc_hh = {};

  // ---- x_t @ {Wr,Wz,Wh}^T  (K = E); B tiles from LDS ----
  const bf16_t* xrow = p.x[dir] + ((size_t)p.t * B_ + (size_t)(mbase + l16)) * E_;
  for (int kb = 0; kb < E_; kb += 32) {
    __builtin_prefetch(xrow + kb + 128);
    v16bf a = ld2x8(xrow + kb + hsel * 16, xrow + kb + hsel * 16 + 8);
    const int bk  = kb + hsel * 16;
    const int idx = lds_idx(l16 * E_ + bk);
    v16bf bv;
    bv = ld2x8(wrS + idx, wrS + idx + 8); WMMA_BF16(acc_r,  a, bv);
    bv = ld2x8(wzS + idx, wzS + idx + 8); WMMA_BF16(acc_z,  a, bv);
    bv = ld2x8(whS + idx, whS + idx + 8); WMMA_BF16(acc_xh, a, bv);
  }

  // ---- h_{t-1} @ {Ur,Uz,Uh}^T  (K = H); Uh kept separate for r*(h@Uh^T) ----
  const bf16_t* hrow = p.hin_b[dir] + (size_t)(mbase + l16) * H_;
  for (int kb = 0; kb < H_; kb += 32) {
    __builtin_prefetch(hrow + kb + 128);
    v16bf a = ld2x8(hrow + kb + hsel * 16, hrow + kb + hsel * 16 + 8);
    const int bk  = kb + hsel * 16;
    const int idx = lds_idx(l16 * H_ + bk);
    v16bf bv;
    bv = ld2x8(urS + idx, urS + idx + 8); WMMA_BF16(acc_r,  a, bv);
    bv = ld2x8(uzS + idx, uzS + idx + 8); WMMA_BF16(acc_z,  a, bv);
    bv = ld2x8(uhS + idx, uhS + idx + 8); WMMA_BF16(acc_hh, a, bv);
  }

  // ---- gates + state update (C/D layout: elem j -> M = j + 8*hsel, N = l16) ----
  const float brn = p.br[dir][n], bzn = p.bz[dir][n], bhn = p.bh[dir][n];
#pragma unroll
  for (int j = 0; j < 8; ++j) {
    const int   bi    = mbase + j + 8 * hsel;     // batch row
    const int   blen  = p.len[bi];
    const float hprev = p.hin_f[dir][(size_t)bi * H_ + n];
    const float r  = 1.0f / (1.0f + __expf(-(acc_r[j] + brn)));
    const float z  = 1.0f / (1.0f + __expf(-(acc_z[j] + bzn)));
    const float hp = tanhf(acc_xh[j] + bhn + r * acc_hh[j]);
    const float hn = z * hp + (1.0f - z) * hprev;
    const bool  valid = p.t < blen;
    const float hc = valid ? hn : hprev;          // frozen carry past length
    p.hout_f[dir][(size_t)bi * H_ + n] = hc;
    p.hout_b[dir][(size_t)bi * H_ + n] = (bf16_t)hc;
    if (dir == 0) {
      p.out[((size_t)p.t * B_ + bi) * (2 * H_) + n] = valid ? hn : 0.0f;
    } else if (valid) {
      const int tp = blen - 1 - p.t;              // un-flip rtl output position
      p.out[((size_t)tp * B_ + bi) * (2 * H_) + H_ + n] = hn;
    }
  }
}

// ---------------------------------------------------------------------------
// dec_init = h_ltr_final @ Ws^T + b   (64 x 1024 x 1024). grid = 64, block = 128.
// ---------------------------------------------------------------------------
__global__ __launch_bounds__(128) void dec_init_wmma(const bf16_t* __restrict__ hfin,
                                                     const bf16_t* __restrict__ wsw,
                                                     const float*  __restrict__ wsb,
                                                     float* __restrict__ outp) {
  const int wave  = threadIdx.x >> 5;
  const int lane  = threadIdx.x & 31;
  const int tile  = blockIdx.x * 4 + wave;
  const int mbase = (tile >> 6) * 16;
  const int nbase = (tile & 63) * 16;
  const int hsel  = lane >> 4;
  const int l16   = lane & 15;
  const int n     = nbase + l16;

  v8f acc = {};
  const bf16_t* hrow = hfin + (size_t)(mbase + l16) * H_;
  const bf16_t* wrow = wsw + (size_t)n * H_;
  for (int kb = 0; kb < H_; kb += 32) {
    v16bf a  = ld2x8(hrow + kb + hsel * 16, hrow + kb + hsel * 16 + 8);
    const int bk = kb + hsel * 16;
    v16bf bv = ld2x8(wrow + bk, wrow + bk + 8);
    WMMA_BF16(acc, a, bv);
  }
  const float bias = wsb[n];
#pragma unroll
  for (int j = 0; j < 8; ++j)
    outp[(size_t)(mbase + j + 8 * hsel) * H_ + n] = acc[j] + bias;
}

// ---------------------------------------------------------------------------
// Host side
// ---------------------------------------------------------------------------
extern "C" void kernel_launch(void* const* d_in, const int* in_sizes, int n_in,
                              void* d_out, int out_size, void* d_ws, size_t ws_size,
                              hipStream_t stream) {
  (void)in_sizes; (void)n_in; (void)out_size; (void)ws_size;
  // Input order (setup_inputs): 0 emb; per dir d: 1+9d..: Wh,Wz,Wr,Uh,Uz,Ur,bh,bz,br;
  // 19 Ws_w; 20 Ws_b; 21 padded_tokens (i32); 22 lengths (i32).
  const float* emb     = (const float*)d_in[0];
  const int*   tokens  = (const int*)d_in[21];
  const int*   lengths = (const int*)d_in[22];
  float*       out     = (float*)d_out;

  uint8_t* cur = (uint8_t*)d_ws;
  auto alloc = [&](size_t bytes) -> void* {
    void* p = (void*)cur;
    cur += (bytes + 255) & ~(size_t)255;
    return p;
  };

  bf16_t* xbuf[2];
  xbuf[0] = (bf16_t*)alloc((size_t)L_ * B_ * E_ * sizeof(bf16_t));
  xbuf[1] = (bf16_t*)alloc((size_t)L_ * B_ * E_ * sizeof(bf16_t));

  bf16_t *whb[2], *wzb[2], *wrb[2], *uhb[2], *uzb[2], *urb[2];
  for (int d = 0; d < 2; ++d) {
    whb[d] = (bf16_t*)alloc((size_t)H_ * E_ * sizeof(bf16_t));
    wzb[d] = (bf16_t*)alloc((size_t)H_ * E_ * sizeof(bf16_t));
    wrb[d] = (bf16_t*)alloc((size_t)H_ * E_ * sizeof(bf16_t));
    uhb[d] = (bf16_t*)alloc((size_t)H_ * H_ * sizeof(bf16_t));
    uzb[d] = (bf16_t*)alloc((size_t)H_ * H_ * sizeof(bf16_t));
    urb[d] = (bf16_t*)alloc((size_t)H_ * H_ * sizeof(bf16_t));
  }
  bf16_t* wsbf = (bf16_t*)alloc((size_t)H_ * H_ * sizeof(bf16_t));

  float*  hf[2][2];
  bf16_t* hb[2][2];
  for (int d = 0; d < 2; ++d)
    for (int ppar = 0; ppar < 2; ++ppar) {
      hf[d][ppar] = (float*)alloc((size_t)B_ * H_ * sizeof(float));
      hb[d][ppar] = (bf16_t*)alloc((size_t)B_ * H_ * sizeof(bf16_t));
    }

  // Weight casts f32 -> bf16
  for (int d = 0; d < 2; ++d) {
    const int base = 1 + 9 * d;
    const float* srcs[6] = {(const float*)d_in[base + 0], (const float*)d_in[base + 1],
                            (const float*)d_in[base + 2], (const float*)d_in[base + 3],
                            (const float*)d_in[base + 4], (const float*)d_in[base + 5]};
    bf16_t* dsts[6] = {whb[d], wzb[d], wrb[d], uhb[d], uzb[d], urb[d]};
    const int ns[6] = {H_ * E_, H_ * E_, H_ * E_, H_ * H_, H_ * H_, H_ * H_};
    for (int i = 0; i < 6; ++i)
      cast_f32_bf16<<<(ns[i] + 255) / 256, 256, 0, stream>>>(srcs[i], dsts[i], ns[i]);
  }
  cast_f32_bf16<<<(H_ * H_ + 255) / 256, 256, 0, stream>>>(
      (const float*)d_in[19], wsbf, H_ * H_);

  // Embedding gather (both directions)
  {
    const size_t total = (size_t)L_ * B_ * E_;
    gather_embed<<<(unsigned)((total + 255) / 256), 256, 0, stream>>>(
        emb, tokens, lengths, xbuf[0], xbuf[1]);
  }

  // Zero-fill hiddens output (rtl padding positions are never written by steps)
  // and the initial hidden-state buffers.
  const size_t hidN = (size_t)L_ * B_ * 2 * H_;
  zero_f32<<<(unsigned)((hidN + 255) / 256), 256, 0, stream>>>(out, hidN);
  for (int d = 0; d < 2; ++d)
    for (int ppar = 0; ppar < 2; ++ppar) {
      zero_f32<<<((B_ * H_) + 255) / 256, 256, 0, stream>>>(hf[d][ppar], (size_t)B_ * H_);
      zero_bf16<<<((B_ * H_) + 255) / 256, 256, 0, stream>>>(hb[d][ppar], (size_t)B_ * H_);
    }

  // Recurrence: 512 dependent step launches (graph replay amortizes launches).
  StepArgs sa;
  for (int d = 0; d < 2; ++d) {
    const int base = 1 + 9 * d;
    sa.x[d]  = xbuf[d];
    sa.Wr[d] = wrb[d]; sa.Wz[d] = wzb[d]; sa.Wh[d] = whb[d];
    sa.Ur[d] = urb[d]; sa.Uz[d] = uzb[d]; sa.Uh[d] = uhb[d];
    sa.bh[d] = (const float*)d_in[base + 6];
    sa.bz[d] = (const float*)d_in[base + 7];
    sa.br[d] = (const float*)d_in[base + 8];
  }
  sa.len = lengths;
  sa.out = out;
  for (int t = 0; t < L_; ++t) {
    const int pi = t & 1, po = pi ^ 1;
    for (int d = 0; d < 2; ++d) {
      sa.hin_f[d]  = hf[d][pi]; sa.hin_b[d]  = hb[d][pi];
      sa.hout_f[d] = hf[d][po]; sa.hout_b[d] = hb[d][po];
    }
    sa.t = t;
    bigru_step_wmma<<<dim3(64, 2, 1), 128, 0, stream>>>(sa);
  }

  // dec_init = final h_ltr (carry frozen past length == h at lengths-1) @ Ws^T + b
  dec_init_wmma<<<64, 128, 0, stream>>>(hb[0][L_ & 1], wsbf,
                                        (const float*)d_in[20], out + hidN);
}
